// CRF_decoder_66039417143495
// MI455X (gfx1250) — matrix operand order repaired and checked by reading, not compiled, not run on hardware
//
#include <hip/hip_runtime.h>
#include <hip/hip_bf16.h>

// CRF decoder for MI455X (gfx1250).
//  Phase A: emission GEMM via v_wmma_f32_16x16x32_f16 (memory-bound)
//  Phase B: forward logsumexp scan (v_exp_f32 bound), chunked 48x48 LSE
//  Phase C: Viterbi max/argmax scan, int8 stage indices
//  Phase D: LDS-staged serial backtrack
// Output layout in d_out: [0..63] loss (f32), [64..65599] path (i32 bit patterns).

#define BATCH   64
#define SEQ     1024
#define DMODEL  512
#define NTAG    48
#define TDIM    50          // transition is (TAG+2)x(TAG+2)
#define WROW    536         // LDS row stride (halves) for W: conflict-free column reads
#define NEG_BIG (-3.0e38f)

typedef __attribute__((ext_vector_type(16))) _Float16 v16h;
typedef __attribute__((ext_vector_type(8)))  _Float16 v8h;
typedef __attribute__((ext_vector_type(8)))  float    v8f;

// ---------------------------------------------------------------------------
// Phase A: tag_vec[bs][t] = dot(F[bs][:], W[t][:]) / sqrt(512)
// One block = 256 threads = 8 waves; each wave owns a 16-row M tile x all 48 tags.
// All three B fragments are fetched from LDS before the first WMMA so a single
// s_wait_dscnt covers three back-to-back v_wmma issues.
// ---------------------------------------------------------------------------
__global__ __launch_bounds__(256)
void emit_gemm_wmma(const float* __restrict__ F, const float* __restrict__ W,
                    float* __restrict__ tagv)
{
    __shared__ _Float16 Wl[NTAG * WROW];

    const int tid = threadIdx.x;
    // Stage W (48x512 f32 -> f16) into LDS once per block.
    for (int idx = tid; idx < NTAG * DMODEL; idx += 256) {
        int t = idx >> 9, k = idx & 511;
        Wl[t * WROW + k] = (_Float16)W[idx];
    }
    __syncthreads();

    const int wave = tid >> 5, lane = tid & 31;
    const int r  = lane & 15;      // M within tile (A) / N within tile (B,C)
    const int kh = lane >> 4;      // K-half selector
    const long row = (long)blockIdx.x * 128 + wave * 16 + r;
    const float* __restrict__ Arow = F + row * DMODEL;

    v8f acc0 = {}, acc1 = {}, acc2 = {};

    for (int kc = 0; kc < DMODEL; kc += 32) {
        if (kc + 32 < DMODEL)
            __builtin_prefetch(Arow + kc + 32, 0, 1);   // global_prefetch_b8

        // A fragment: contiguous K order, same as B. Lane holds K = kc+kh*16+[0..15]
        // as elements 0..15.
        const float4 f0 = *(const float4*)(Arow + kc + kh * 16);
        const float4 f1 = *(const float4*)(Arow + kc + kh * 16 + 4);
        const float4 f2 = *(const float4*)(Arow + kc + kh * 16 + 8);
        const float4 f3 = *(const float4*)(Arow + kc + kh * 16 + 12);

        // B fragments from LDS: B[k][n] = W[n][k]. Lane n=r, K range kc+kh*16+[0..15].
        // Issue all six ds_load_b128 up front.
        const _Float16* bp0 = &Wl[( 0 + r) * WROW + kc + kh * 16];
        const _Float16* bp1 = &Wl[(16 + r) * WROW + kc + kh * 16];
        const _Float16* bp2 = &Wl[(32 + r) * WROW + kc + kh * 16];
        v8h b0lo = *(const v8h*)bp0;       v8h b0hi = *(const v8h*)(bp0 + 8);
        v8h b1lo = *(const v8h*)bp1;       v8h b1hi = *(const v8h*)(bp1 + 8);
        v8h b2lo = *(const v8h*)bp2;       v8h b2hi = *(const v8h*)(bp2 + 8);

        v16h a;
        a[0]=(_Float16)f0.x; a[1]=(_Float16)f0.y; a[2]=(_Float16)f0.z; a[3]=(_Float16)f0.w;
        a[4]=(_Float16)f1.x; a[5]=(_Float16)f1.y; a[6]=(_Float16)f1.z; a[7]=(_Float16)f1.w;
        a[8]=(_Float16)f2.x; a[9]=(_Float16)f2.y; a[10]=(_Float16)f2.z; a[11]=(_Float16)f2.w;
        a[12]=(_Float16)f3.x; a[13]=(_Float16)f3.y; a[14]=(_Float16)f3.z; a[15]=(_Float16)f3.w;

        v16h bv0, bv1, bv2;
        #pragma unroll
        for (int q = 0; q < 8; ++q) {
            bv0[q] = b0lo[q]; bv0[q + 8] = b0hi[q];
            bv1[q] = b1lo[q]; bv1[q + 8] = b1hi[q];
            bv2[q] = b2lo[q]; bv2[q + 8] = b2hi[q];
        }

        acc0 = __builtin_amdgcn_wmma_f32_16x16x32_f16(false, a, false, bv0,
                                                      (short)0, acc0, false, false);
        acc1 = __builtin_amdgcn_wmma_f32_16x16x32_f16(false, a, false, bv1,
                                                      (short)0, acc1, false, false);
        acc2 = __builtin_amdgcn_wmma_f32_16x16x32_f16(false, a, false, bv2,
                                                      (short)0, acc2, false, false);
    }

    // C layout: VGPR rr -> M = rr (lanes 0-15) or rr+8 (lanes 16-31); N = lane&15.
    const float scale = 0.044194173824159216f;   // 1/sqrt(512)
    const long mbase = (long)blockIdx.x * 128 + wave * 16 + kh * 8;
    #pragma unroll
    for (int rr = 0; rr < 8; ++rr) {
        tagv[(mbase + rr) * NTAG +  0 + r] = acc0[rr] * scale;
        tagv[(mbase + rr) * NTAG + 16 + r] = acc1[rr] * scale;
        tagv[(mbase + rr) * NTAG + 32 + r] = acc2[rr] * scale;
    }
}

// ---------------------------------------------------------------------------
// Phase B: forward logsumexp scan + gold-path score -> loss.
// One block per batch; 192 threads = (j = tid%48) x (chunk c = tid/48, 12 i's each).
// ---------------------------------------------------------------------------
__global__ __launch_bounds__(192)
void crf_forward(const float* __restrict__ tagv, const int* __restrict__ y,
                 const unsigned char* __restrict__ mask,
                 const float* __restrict__ trans,
                 float* __restrict__ loss_out, int* __restrict__ Lws)
{
    __shared__ float transT[NTAG * 49];   // transT[j][i] = trans_tt[i][j]
    __shared__ float startv[NTAG], endv[NTAG];
    __shared__ float aBuf[NTAG], eBuf[NTAG];
    __shared__ float pmax[4 * NTAG], psum[4 * NTAG];
    __shared__ float bm[SEQ];
    __shared__ float prevFin[NTAG];
    __shared__ int   red[192];
    __shared__ int   Lsh;

    const int b = blockIdx.x, tid = threadIdx.x;
    const int j = tid % NTAG, c = tid / NTAG;
    const long ybase = (long)b * SEQ;

    for (int idx = tid; idx < NTAG * NTAG; idx += 192) {
        int i = idx / NTAG, jj = idx % NTAG;
        transT[jj * 49 + i] = trans[i * TDIM + jj];
    }
    if (tid < NTAG) {
        startv[tid] = trans[NTAG * TDIM + tid];
        endv[tid]   = trans[tid * TDIM + (NTAG + 1)];
    }
    int msum = 0;
    for (int s = tid; s < SEQ; s += 192) {
        unsigned char m = mask[ybase + s];
        bm[s] = m ? 0.0f : 1.0f;
        msum += m;
    }
    red[tid] = msum;
    __syncthreads();
    if (tid == 0) {
        int tot = 0;
        for (int t = 0; t < 192; ++t) tot += red[t];
        Lsh = SEQ - tot - 1;
    }
    __syncthreads();

    float prev_j = (c == 0) ? startv[j] : 0.0f;
    float score  = startv[y[ybase]];     // meaningful on tid 0 only
    int   yprev  = y[ybase];
    const int Lb = Lsh;

    for (int s = 0; s < SEQ; ++s) {
        const float m0 = bm[s];
        const float m1 = (s < SEQ - 1) ? bm[s + 1] : 0.0f;

        __syncthreads();                                   // B1
        if (c == 0) {
            float e = tagv[(ybase + s) * NTAG + j];
            eBuf[j] = e;
            aBuf[j] = prev_j + e * m0;
        }
        __syncthreads();                                   // B2

        float v[12];
        float pm = NEG_BIG;
        const int ib = c * 12;
        #pragma unroll
        for (int t = 0; t < 12; ++t) {
            float val = aBuf[ib + t] + transT[j * 49 + ib + t] * m1;
            v[t] = val;
            pm = fmaxf(pm, val);
        }
        pmax[c * NTAG + j] = pm;
        __syncthreads();                                   // B3

        const float gm = fmaxf(fmaxf(pmax[j], pmax[NTAG + j]),
                               fmaxf(pmax[2 * NTAG + j], pmax[3 * NTAG + j]));
        float ps = 0.0f;
        #pragma unroll
        for (int t = 0; t < 12; ++t) ps += __expf(v[t] - gm);
        psum[c * NTAG + j] = ps;
        __syncthreads();                                   // B4

        if (c == 0) {
            prev_j = gm + __logf(psum[j] + psum[NTAG + j] +
                                 psum[2 * NTAG + j] + psum[3 * NTAG + j]);
        }
        if (tid == 0) {
            int ys = y[ybase + s];
            score += eBuf[ys] * m0;
            if (s > 0) score += trans[yprev * TDIM + ys] * m0;
            yprev = ys;
        }
    }

    if (c == 0) prevFin[j] = prev_j + endv[j];
    __syncthreads();
    if (tid == 0) {
        float gm = NEG_BIG;
        for (int t = 0; t < NTAG; ++t) gm = fmaxf(gm, prevFin[t]);
        float sum = 0.0f;
        for (int t = 0; t < NTAG; ++t) sum += __expf(prevFin[t] - gm);
        float logZ = gm + __logf(sum);
        score += endv[y[ybase + Lb]];
        loss_out[b] = logZ - score;
        Lws[b] = Lb;
    }
}

// ---------------------------------------------------------------------------
// Phase C: Viterbi scan. Same chunked structure; argmax with first-max ties.
// stage_idx stored as int8 (tags < 48) to keep backtrack data L2-resident.
// ---------------------------------------------------------------------------
__global__ __launch_bounds__(192)
void crf_viterbi(const float* __restrict__ tagv, const float* __restrict__ trans,
                 const int* __restrict__ Lws,
                 unsigned char* __restrict__ stage_idx, int* __restrict__ now0_out)
{
    __shared__ float transT[NTAG * 49];
    __shared__ float startv[NTAG], endv[NTAG];
    __shared__ float bBuf[NTAG];
    __shared__ float pmax[4 * NTAG];
    __shared__ int   pidx[4 * NTAG];
    __shared__ float finBuf[NTAG];

    const int b = blockIdx.x, tid = threadIdx.x;
    const int j = tid % NTAG, c = tid / NTAG;
    const long base = (long)b * SEQ;

    for (int idx = tid; idx < NTAG * NTAG; idx += 192) {
        int i = idx / NTAG, jj = idx % NTAG;
        transT[jj * 49 + i] = trans[i * TDIM + jj];
    }
    if (tid < NTAG) {
        startv[tid] = trans[NTAG * TDIM + tid];
        endv[tid]   = trans[tid * TDIM + (NTAG + 1)];
    }
    __syncthreads();

    const int Lb = Lws[b];
    float best_j = 0.0f;
    if (c == 0) {
        best_j = startv[j] + tagv[base * NTAG + j];   // s = 0
        if (Lb == 0) finBuf[j] = best_j;
    }

    for (int s = 1; s < SEQ; ++s) {
        __syncthreads();                               // V1
        if (c == 0) bBuf[j] = best_j;
        __syncthreads();                               // V2

        float pm = NEG_BIG;
        int   pi = 0;
        const int ib = c * 12;
        #pragma unroll
        for (int t = 0; t < 12; ++t) {
            float val = bBuf[ib + t] + transT[j * 49 + ib + t];
            if (val > pm) { pm = val; pi = ib + t; }
        }
        pmax[c * NTAG + j] = pm;
        pidx[c * NTAG + j] = pi;
        __syncthreads();                               // V3

        if (c == 0) {
            float m = pmax[j];
            int  ii = pidx[j];
            #pragma unroll
            for (int cc = 1; cc < 4; ++cc) {
                float mc = pmax[cc * NTAG + j];
                if (mc > m) { m = mc; ii = pidx[cc * NTAG + j]; }
            }
            best_j = m + tagv[(base + s) * NTAG + j];
            stage_idx[(base + s) * NTAG + j] = (unsigned char)ii;
            if (s == Lb) finBuf[j] = best_j;
        }
    }

    __syncthreads();
    if (tid == 0) {
        float m = NEG_BIG;
        int arg = 0;
        for (int t = 0; t < NTAG; ++t) {
            float v = finBuf[t] + endv[t];
            if (v > m) { m = v; arg = t; }
        }
        now0_out[b] = arg;
    }
}

// ---------------------------------------------------------------------------
// Phase D: backtrack. Coalesce 128-step map tiles into LDS, then one lane
// walks the dependent chain at LDS latency instead of L2 latency.
// ---------------------------------------------------------------------------
__global__ __launch_bounds__(128)
void crf_backtrack(const unsigned char* __restrict__ stage_idx,
                   const int* __restrict__ now0_in, const int* __restrict__ Lws,
                   int* __restrict__ path_out)
{
    const int TILE = 128;
    __shared__ unsigned char tile[TILE * NTAG];
    __shared__ int outb[TILE];
    __shared__ int nowSh;

    const int b = blockIdx.x, tid = threadIdx.x;
    const int Lb = Lws[b];
    const int n0 = now0_in[b];
    int* __restrict__ path = path_out + b * SEQ;
    const long base = (long)b * SEQ;

    for (int p = tid; p < SEQ; p += 128)
        if (p > Lb) path[p] = -1;
    if (tid == 0) { path[Lb] = n0; nowSh = n0; }

    for (int jhi = Lb; jhi >= 1; jhi -= TILE) {
        int jlo = jhi - TILE + 1;
        if (jlo < 1) jlo = 1;
        const int cnt = jhi - jlo + 1;

        for (int idx = tid; idx < cnt * NTAG; idx += 128)
            tile[idx] = stage_idx[(base + jlo) * NTAG + idx];
        __syncthreads();

        if (tid == 0) {
            int now = nowSh;
            for (int jj = jhi; jj >= jlo; --jj) {
                int prev = tile[(jj - jlo) * NTAG + now];
                outb[jj - jlo] = prev;
                now = prev;
            }
            nowSh = now;
        }
        __syncthreads();

        for (int idx = tid; idx < cnt; idx += 128)
            path[jlo - 1 + idx] = outb[idx];
        __syncthreads();
    }
}

// ---------------------------------------------------------------------------
extern "C" void kernel_launch(void* const* d_in, const int* in_sizes, int n_in,
                              void* d_out, int out_size, void* d_ws, size_t ws_size,
                              hipStream_t stream)
{
    const float*         F     = (const float*)d_in[0];          // (64,1024,512) f32
    const int*           y     = (const int*)d_in[1];            // (64,1024)     i32
    const unsigned char* mask  = (const unsigned char*)d_in[2];  // (64,1024)     bool
    const float*         W     = (const float*)d_in[3];          // (48,512)      f32
    const float*         trans = (const float*)d_in[4];          // (50,50)       f32

    char* ws = (char*)d_ws;
    float* tagv = (float*)ws;               ws += (size_t)BATCH * SEQ * NTAG * sizeof(float);
    int*   Lws  = (int*)ws;                 ws += BATCH * sizeof(int);
    int*   now0 = (int*)ws;                 ws += BATCH * sizeof(int);
    unsigned char* sidx = (unsigned char*)ws; ws += (size_t)BATCH * SEQ * NTAG;

    float* loss = (float*)d_out;
    int*   path = (int*)d_out + BATCH;

    emit_gemm_wmma<<<(BATCH * SEQ) / 128, 256, 0, stream>>>(F, W, tagv);
    crf_forward   <<<BATCH, 192, 0, stream>>>(tagv, y, mask, trans, loss, Lws);
    crf_viterbi   <<<BATCH, 192, 0, stream>>>(tagv, trans, Lws, sidx, now0);
    crf_backtrack <<<BATCH, 128, 0, stream>>>(sidx, now0, Lws, path);
}
